// WindowAttention_64080912056606
// MI455X (gfx1250) — matrix operand order repaired and checked by reading, not compiled, not run on hardware
//
#include <hip/hip_runtime.h>
#include <hip/hip_bf16.h>
#include <stdint.h>

// Swin window-attention, fused, f16 WMMA pipeline for gfx1250 (MI455X).
// Bn=2048, N=49, C=512, H=16, D=32, nW=64, WIN=7, SIZE=56, SHIFT=3.

typedef __attribute__((ext_vector_type(16))) _Float16 v16h;
typedef __attribute__((ext_vector_type(8)))  float    v8f;

#define LDS_STRIDE 40   // padded row stride in halves (80B: 16B-aligned, few bank conflicts)

union Frag {
    v16h h;
    uint4 q[2];
};

// A-matrix fragment (16x32 f16): lane holds row M = lane&15; its 16 halves
// are K = 16*hi..16*hi+15.
__device__ __forceinline__ v16h ld_frag_a(const _Float16* rowbase, int hi) {
    Frag f;
    f.q[0] = *(const uint4*)(rowbase + 16 * hi);
    f.q[1] = *(const uint4*)(rowbase + 16 * hi + 8);
    return f.h;
}

// B-matrix fragment (32x16 f16, KxN): lane holds column N = lane&15; its 16
// halves are K = 16*hi..16*hi+15 (B stored K-contiguous / transposed).
__device__ __forceinline__ v16h ld_frag_b(const _Float16* colbase, int hi) {
    const uint4* p = (const uint4*)(colbase + 16 * hi);
    Frag f;
    f.q[0] = p[0];
    f.q[1] = p[1];
    return f.h;
}

__device__ __forceinline__ v16h zero_frag() {
    Frag f;
    f.q[0] = make_uint4(0, 0, 0, 0);
    f.q[1] = make_uint4(0, 0, 0, 0);
    return f.h;
}

// ---------------------------------------------------------------------------
// Kernel 0: fp32 -> f16 conversion of x (once; keeps GEMM passes in L2).
// ---------------------------------------------------------------------------
__global__ __launch_bounds__(256) void cvt_f16_kernel(
    const float* __restrict__ x, _Float16* __restrict__ xh) {
    size_t i4 = ((size_t)blockIdx.x * 256 + threadIdx.x) * 4;
    float4 v = *(const float4*)(x + i4);
    union { _Float16 h[4]; uint2 u; } o;
    o.h[0] = (_Float16)v.x;
    o.h[1] = (_Float16)v.y;
    o.h[2] = (_Float16)v.z;
    o.h[3] = (_Float16)v.w;
    *(uint2*)(xh + i4) = o.u;
}

// ---------------------------------------------------------------------------
// Shared GEMM mainloop: C[128 x 64] tile, A f16 [M x 512] row-major,
// B fp32 [512 x ldb].  256 threads = 8 waves; wave w owns rows 16w..16w+15
// and all 64 columns (4 accumulators).  K = 512 in 16 steps of 32.
// ---------------------------------------------------------------------------
__device__ __forceinline__ void gemm_mainloop(
    const _Float16* __restrict__ A, const float* __restrict__ B,
    int m0, int n0, int ldb,
    _Float16* As, _Float16* Bs, v8f acc[4]) {
    const int tid = threadIdx.x;
    const int lane = tid & 31;
    const int wv = tid >> 5;
    const int hi = lane >> 4;
    const int ln = lane & 15;

    for (int kt = 0; kt < 16; ++kt) {
        // Stage A: 128 rows x 32 halves via b128 (8 halves per op, 2 iters)
        const _Float16* as = A + (size_t)m0 * 512 + kt * 32;
        #pragma unroll
        for (int it = 0; it < 2; ++it) {
            int e = it * 256 + tid;           // 0..511
            int row = e >> 2;
            int c8 = (e & 3) * 8;
            *(uint4*)&As[row * LDS_STRIDE + c8] =
                *(const uint4*)(as + (size_t)row * 512 + c8);
        }
        // Stage B transposed (k-pairs packed to one b32 store):
        // rows kt*32..+31, cols n0..n0+63; coalesced in n.
        #pragma unroll
        for (int it = 0; it < 4; ++it) {
            int e = it * 256 + tid;           // 0..1023
            int k = (e >> 6) * 2;
            int n = e & 63;
            const float* bp = B + (size_t)(kt * 32 + k) * ldb + n0 + n;
            union { _Float16 h[2]; uint32_t u; } p;
            p.h[0] = (_Float16)bp[0];
            p.h[1] = (_Float16)bp[ldb];
            *(uint32_t*)&Bs[n * LDS_STRIDE + k] = p.u;
        }
        // Prefetch next K-tile into caches (global_prefetch_b8)
        if (kt < 15) {
            __builtin_prefetch(as + (size_t)(tid >> 1) * 512 + 32 + (tid & 1) * 16, 0, 1);
            __builtin_prefetch(&B[(size_t)((kt + 1) * 32 + (tid >> 6)) * ldb + n0 + (tid & 63)], 0, 1);
        }
        __syncthreads();

        v16h a = ld_frag_a(&As[(16 * wv + ln) * LDS_STRIDE], hi);
        #pragma unroll
        for (int t = 0; t < 4; ++t) {
            v16h b = ld_frag_b(&Bs[(16 * t + ln) * LDS_STRIDE], hi);
            acc[t] = __builtin_amdgcn_wmma_f32_16x16x32_f16(
                false, a, false, b, (short)0, acc[t], false, false);
        }
        __syncthreads();
    }
}

// ---------------------------------------------------------------------------
// Kernel 1: QKV projection.  xh[100352 x 512] * qkv_w[512 x 1536] + bias.
// Writes q (scaled by D^-0.5), k, v as f16 with layout [b*16+h][49][32].
// ---------------------------------------------------------------------------
__global__ __launch_bounds__(256) void qkv_gemm_kernel(
    const _Float16* __restrict__ xh, const float* __restrict__ w,
    const float* __restrict__ bias,
    _Float16* __restrict__ qb, _Float16* __restrict__ kb,
    _Float16* __restrict__ vb) {
    __shared__ _Float16 As[128 * LDS_STRIDE];
    __shared__ _Float16 Bs[64 * LDS_STRIDE];

    const int m0 = blockIdx.x * 128;
    const int n0 = blockIdx.y * 64;
    const int lane = threadIdx.x & 31;
    const int wv = threadIdx.x >> 5;
    const int hi = lane >> 4;
    const int ln = lane & 15;

    v8f acc[4] = {{}, {}, {}, {}};
    gemm_mainloop(xh, w, m0, n0, 1536, As, Bs, acc);

    // One division, then incremental (b, tok) per output row.
    const int base = m0 + 16 * wv + 8 * hi;
    int bb = base / 49;
    int tt = base - bb * 49;
    int brow[8], trow[8];
    #pragma unroll
    for (int r = 0; r < 8; ++r) {
        brow[r] = bb;
        trow[r] = tt;
        if (++tt == 49) { tt = 0; ++bb; }
    }
    #pragma unroll
    for (int t = 0; t < 4; ++t) {
        const int c = n0 + 16 * t + ln;       // global column (0..1535)
        const int three = c >> 9;             // 0=q 1=k 2=v
        const int h = (c >> 5) & 15;
        const int d = c & 31;
        const float bv = bias[c];
        const float scale = (three == 0) ? 0.17677669529663687f : 1.0f;
        _Float16* dst = (three == 0) ? qb : ((three == 1) ? kb : vb);
        #pragma unroll
        for (int r = 0; r < 8; ++r) {
            dst[(((size_t)brow[r] * 16 + h) * 49 + trow[r]) * 32 + d] =
                (_Float16)((acc[t][r] + bv) * scale);
        }
    }
}

// ---------------------------------------------------------------------------
// Kernel 2: fused attention per (b, h): scores + bias + shifted-window mask
// + softmax + P@V.  Block = 128 threads (4 waves).  49 padded to 64.
// ---------------------------------------------------------------------------
__device__ __forceinline__ int region1(int p) {
    return (p < 49) ? 0 : ((p < 53) ? 1 : 2);
}

__global__ __launch_bounds__(128) void attn_kernel(
    const _Float16* __restrict__ qb, const _Float16* __restrict__ kb,
    const _Float16* __restrict__ vb, const float* __restrict__ bias_table,
    _Float16* __restrict__ ob) {
    __shared__ float    S[64 * 68];     // scores (padded row stride)
    __shared__ float    biasS[2401];    // bias[h][49][49]
    __shared__ _Float16 P[64 * 64];     // softmax probs, f16, zero-padded
    __shared__ _Float16 vT[32 * 64];    // v transposed [d][j], zero-padded

    const int bh = blockIdx.x;
    const int b = bh >> 4;
    const int h = bh & 15;
    const int tid = threadIdx.x;
    const int lane = tid & 31;
    const int wv = tid >> 5;
    const int hi = lane >> 4;
    const int ln = lane & 15;

    const _Float16* qh = qb + (size_t)bh * 49 * 32;
    const _Float16* kh = kb + (size_t)bh * 49 * 32;
    const _Float16* vh = vb + (size_t)bh * 49 * 32;

    // Stage relative-position bias for this head (computed index, gathered).
    for (int e = tid; e < 2401; e += 128) {
        int i = e / 49;
        int j = e - i * 49;
        int idx = ((i / 7 - j / 7) + 6) * 13 + ((i % 7 - j % 7) + 6);
        biasS[e] = bias_table[idx * 16 + h];
    }
    // Stage v transposed with zero padding for j >= 49.
    for (int e = tid; e < 32 * 64; e += 128) {
        int d = e >> 6;
        int j = e & 63;
        vT[d * 64 + j] = (j < 49) ? vh[j * 32 + d] : (_Float16)0.0f;
    }
    __syncthreads();

    // ---- scores: S = q @ k^T  (per wave: one 16-row strip, 4 col tiles) ----
    const int mrowA = 16 * wv + ln;
    v16h aq = (mrowA < 49) ? ld_frag_a(qh + mrowA * 32, hi) : zero_frag();

    v8f sc[4];
    #pragma unroll
    for (int t = 0; t < 4; ++t) {
        int n = 16 * t + ln;
        v16h bk = (n < 49) ? ld_frag_b(kh + n * 32, hi) : zero_frag();
        v8f z = {};
        sc[t] = __builtin_amdgcn_wmma_f32_16x16x32_f16(false, aq, false, bk,
                                                       (short)0, z, false, false);
    }

    // Epilogue: + bias + shifted-window mask, store to LDS.
    const int wid = b & 63;           // window index within image
    const int wr = wid >> 3;
    const int wc = wid & 7;
    const int mr = 16 * wv + 8 * hi;  // first row this lane holds
    #pragma unroll
    for (int t = 0; t < 4; ++t) {
        #pragma unroll
        for (int r = 0; r < 8; ++r) {
            int row = mr + r;
            int cl = 16 * t + ln;
            float v = sc[t][r];
            if (row < 49 && cl < 49) {
                v += biasS[row * 49 + cl];
                int ri = region1(wr * 7 + row / 7) * 3 + region1(wc * 7 + row % 7);
                int rj = region1(wr * 7 + cl / 7) * 3 + region1(wc * 7 + cl % 7);
                if (ri != rj) v -= 100.0f;
            } else {
                v = -1e30f;
            }
            S[row * 68 + cl] = v;
        }
    }
    __syncthreads();

    // ---- softmax (one thread per row), produce f16 P with zero padding ----
    if (tid < 64) {
        if (tid < 49) {
            float* row = &S[tid * 68];
            float mx = -1e30f;
            for (int j = 0; j < 49; ++j) mx = fmaxf(mx, row[j]);
            float sum = 0.0f;
            for (int j = 0; j < 49; ++j) {
                float ex = __expf(row[j] - mx);
                row[j] = ex;
                sum += ex;
            }
            float inv = 1.0f / sum;
            for (int j = 0; j < 64; ++j)
                P[tid * 64 + j] = (j < 49) ? (_Float16)(row[j] * inv)
                                           : (_Float16)0.0f;
        } else {
            for (int j = 0; j < 64; ++j) P[tid * 64 + j] = (_Float16)0.0f;
        }
    }
    __syncthreads();

    // ---- O = P @ v : K = 64 (2 steps), N = 32 (2 tiles) per wave ----------
    v8f oacc[2] = {{}, {}};
    #pragma unroll
    for (int s = 0; s < 2; ++s) {
        v16h ap = ld_frag_a(&P[(16 * wv + ln) * 64 + 32 * s], hi);
        #pragma unroll
        for (int nt = 0; nt < 2; ++nt) {
            v16h bv2 = ld_frag_b(&vT[(16 * nt + ln) * 64 + 32 * s], hi);
            oacc[nt] = __builtin_amdgcn_wmma_f32_16x16x32_f16(
                false, ap, false, bv2, (short)0, oacc[nt], false, false);
        }
    }

    // Store o as f16 in [token][h*32+d] layout (ready for proj GEMM).
    #pragma unroll
    for (int nt = 0; nt < 2; ++nt) {
        #pragma unroll
        for (int r = 0; r < 8; ++r) {
            int row = mr + r;
            if (row < 49) {
                int d = 16 * nt + ln;
                ob[((size_t)b * 49 + row) * 512 + h * 32 + d] = (_Float16)oacc[nt][r];
            }
        }
    }
}

// ---------------------------------------------------------------------------
// Kernel 3: output projection.  ob[100352 x 512](f16) * proj_w[512 x 512]
// + bias -> fp32 out.
// ---------------------------------------------------------------------------
__global__ __launch_bounds__(256) void proj_gemm_kernel(
    const _Float16* __restrict__ a, const float* __restrict__ w,
    const float* __restrict__ bias, float* __restrict__ out) {
    __shared__ _Float16 As[128 * LDS_STRIDE];
    __shared__ _Float16 Bs[64 * LDS_STRIDE];

    const int m0 = blockIdx.x * 128;
    const int n0 = blockIdx.y * 64;
    const int lane = threadIdx.x & 31;
    const int wv = threadIdx.x >> 5;
    const int hi = lane >> 4;
    const int ln = lane & 15;

    v8f acc[4] = {{}, {}, {}, {}};
    gemm_mainloop(a, w, m0, n0, 512, As, Bs, acc);

    #pragma unroll
    for (int t = 0; t < 4; ++t) {
        const int c = n0 + 16 * t + ln;
        const float bv = bias[c];
        #pragma unroll
        for (int r = 0; r < 8; ++r) {
            int m = m0 + 16 * wv + r + 8 * hi;
            out[(size_t)m * 512 + c] = acc[t][r] + bv;
        }
    }
}

// ---------------------------------------------------------------------------
extern "C" void kernel_launch(void* const* d_in, const int* in_sizes, int n_in,
                              void* d_out, int out_size, void* d_ws, size_t ws_size,
                              hipStream_t stream) {
    const float* x          = (const float*)d_in[0];
    const float* qkv_w      = (const float*)d_in[1];
    const float* qkv_b      = (const float*)d_in[2];
    const float* proj_w     = (const float*)d_in[3];
    const float* proj_b     = (const float*)d_in[4];
    const float* bias_table = (const float*)d_in[5];
    float* out = (float*)d_out;

    // Workspace: q, k, v, o, xh -- each 100352*512 = 51,380,224 f16 (~98 MB).
    const size_t BUF_HALVES = (size_t)100352 * 512;
    _Float16* qb = (_Float16*)d_ws;
    _Float16* kb = qb + BUF_HALVES;
    _Float16* vb = kb + BUF_HALVES;
    _Float16* ob = vb + BUF_HALVES;   // [100352][512] f16
    _Float16* xh = ob + BUF_HALVES;   // x converted to f16 (L2-resident)

    // 0) Convert x to f16 once (51.4M elems, 4 per thread).
    cvt_f16_kernel<<<dim3(50176), 256, 0, stream>>>(x, xh);
    // 1) QKV projection: M=100352 (784 tiles of 128), N=1536 (24 tiles of 64)
    qkv_gemm_kernel<<<dim3(784, 24), 256, 0, stream>>>(xh, qkv_w, qkv_b,
                                                       qb, kb, vb);
    // 2) Fused attention per (b, h)
    attn_kernel<<<dim3(2048 * 16), 128, 0, stream>>>(qb, kb, vb, bias_table, ob);
    // 3) Output projection: N=512 (8 tiles of 64)
    proj_gemm_kernel<<<dim3(784, 8), 256, 0, stream>>>(ob, proj_w, proj_b, out);
}
